// Net_88923002896659
// MI455X (gfx1250) — compile-verified, matrix operand order checked by reading, not run on hardware
//
#include <hip/hip_runtime.h>
#include <hip/hip_bf16.h>
#include <stdint.h>

// ---------------------------------------------------------------------------
// MI455X (gfx1250) implementation.
//  - All large matmuls via v_wmma_f32_16x16x32_f16 (f16 in, f32 accum).
//  - B matrices pre-transposed (NxK) so both A and B fragments are contiguous
//    ds_load_b128 from LDS (no strided u16 gather).
//  - Global->LDS staging via Tensor Data Mover (tensor_load_to_lds), double
//    buffered with s_wait_tensorcnt; TDM pad fields give conflict-free LDS.
//  - FFT/IFFT as dense DFT GEMMs (twiddle matrices are symmetric == own
//    transpose); pipeline is compute-bound so matrix pipes do all heavy math.
// ---------------------------------------------------------------------------

typedef _Float16 v16h __attribute__((ext_vector_type(16)));
typedef float    v8f  __attribute__((ext_vector_type(8)));
typedef unsigned int v4u __attribute__((ext_vector_type(4)));
typedef unsigned int v8u __attribute__((ext_vector_type(8)));

#define NB    256
#define NK    4
#define NCO   8
#define NS    2400
#define RR    40
#define DD    512
#define NLAY  4
#define NCLS  10
#define TOK   75
#define NHD   4
#define DH    128

// ------------------------------ device helpers -----------------------------

__device__ __forceinline__ float gelu_f(float x) {
  return 0.5f * x * (1.0f + erff(x * 0.70710678118f));
}

__device__ __forceinline__ unsigned lds_off32(const void* p) {
  // generic (flat) LDS address: low 32 bits == LDS byte offset
  return (unsigned)(unsigned long long)p;
}

__device__ __forceinline__ v8f wmma_f16(v16h a, v16h b, v8f c) {
  return __builtin_amdgcn_wmma_f32_16x16x32_f16(false, a, false, b,
                                                (short)0, c, false, false);
}

// 16x32 f16 fragment, rows contiguous in LDS with stride `ld` halves.
// Used for A and (because B is stored transposed NxK) for B as well.
__device__ __forceinline__ v16h frag_ld(const _Float16* s, int ld, int lane) {
  int r = lane & 15, hi = lane >> 4;
  const _Float16* p = s + r * ld + hi * 8;
  v16h f;
#pragma unroll
  for (int i = 0; i < 8; ++i) { f[i] = p[i]; f[8 + i] = p[16 + i]; }
  return f;
}

// Tensor Data Mover: 2D tile load global->LDS (2-group descriptor form).
// dims/strides in elements (data_size = 2 bytes). OOB rows/cols zero-filled.
__device__ __forceinline__ void tdm_load2d(unsigned lds_addr, const void* g,
                                           unsigned dim0, unsigned dim1,
                                           unsigned stride0,
                                           unsigned tile0, unsigned tile1,
                                           unsigned pad_int, unsigned pad_amt,
                                           unsigned pad_en) {
  unsigned long long ga = (unsigned long long)g;
  v4u g0;
  g0[0] = 1u;                                  // count=1, user descriptor
  g0[1] = lds_addr;                            // LDS byte address
  g0[2] = (unsigned)ga;                        // global_addr[31:0]
  g0[3] = (unsigned)((ga >> 32) & 0x1FFFFFFull) | (2u << 30);  // addr hi | type=2
  v8u g1;
  g1[0] = (1u << 16) | (pad_en << 20) | (pad_int << 22) | (pad_amt << 25);
  g1[1] = (dim0 & 0xFFFFu) << 16;              // tensor_dim0 lo16
  g1[2] = (dim0 >> 16) | ((dim1 & 0xFFFFu) << 16);
  g1[3] = (dim1 >> 16) | (tile0 << 16);
  g1[4] = tile1 & 0xFFFFu;                     // tile_dim1 (tile_dim2=0)
  g1[5] = stride0;                             // tensor_dim0_stride lo32
  g1[6] = 0u;
  g1[7] = 0u;
  asm volatile("tensor_load_to_lds %0, %1" :: "s"(g0), "s"(g1) : "memory");
}

// ------------------------------ generic GEMM -------------------------------
// C[M,N] = act(alpha*A[M,K]@Bt[N,K]^T + beta*Cin + bias)   (A,Bt f16)
// Block tile 128x128, K-step 32; 8 waves, each 32x64 (2x4 WMMA tiles):
// per k-step per wave: 6 fragment loads (12 ds_load_b128) feed 8 WMMAs.

#define GBM 128
#define GBN 128
#define GBK 32
#define LDT_S 40   // 32 halves + TDM pad (4 dwords per 16 dwords)

__launch_bounds__(256)
__global__ void k_gemm(const _Float16* __restrict__ A, int lda,
                       const _Float16* __restrict__ Bt, int ldbt,
                       const float* __restrict__ bias,
                       const float* __restrict__ Cin,
                       float* __restrict__ outF,
                       _Float16* __restrict__ outH, int ldc,
                       int M, int N, int Kd,
                       float alpha, float beta, int act) {
  __shared__ __align__(16) _Float16 sA[2][GBM * LDT_S];
  __shared__ __align__(16) _Float16 sB[2][GBN * LDT_S];
  const int tid = threadIdx.x;
  const int wid = tid >> 5;
  const int lane = tid & 31;
  const int m0 = blockIdx.y * GBM;
  const int n0 = blockIdx.x * GBN;
  const int wr = wid >> 1, wc = wid & 1;

  const unsigned aoff = lds_off32(&sA[0][0]);
  const unsigned boff = lds_off32(&sB[0][0]);
  const unsigned tbytes = GBM * LDT_S * 2;

  v8f acc[2][4] = {};
  const int nk = Kd / GBK;

  auto issue = [&](int kt, int buf) {
    const _Float16* ga = A + (size_t)m0 * lda + (size_t)kt * GBK;
    tdm_load2d(aoff + (unsigned)buf * tbytes, ga,
               (unsigned)(Kd - kt * GBK), (unsigned)(M - m0), (unsigned)lda,
               GBK, GBM, /*pad: per 16 dwords*/3, /*4 dwords*/3, 1);
    const _Float16* gb = Bt + (size_t)n0 * ldbt + (size_t)kt * GBK;
    tdm_load2d(boff + (unsigned)buf * tbytes, gb,
               (unsigned)(Kd - kt * GBK), (unsigned)(N - n0), (unsigned)ldbt,
               GBK, GBN, 3, 3, 1);
  };

  if (wid == 0 && nk > 0) issue(0, 0);

  for (int kt = 0; kt < nk; ++kt) {
    const int buf = kt & 1;
    if (wid == 0) {
      if (kt + 1 < nk) {
        issue(kt + 1, buf ^ 1);
        __builtin_amdgcn_s_wait_tensorcnt(2);   // current buffer's 2 loads done
      } else {
        __builtin_amdgcn_s_wait_tensorcnt(0);   // last iteration: drain all
      }
    }
    __syncthreads();
    const _Float16* As = &sA[buf][0] + (wr * 32) * LDT_S;
    const _Float16* Bs = &sB[buf][0] + (wc * 64) * LDT_S;
    v16h bfr[4];
#pragma unroll
    for (int ni = 0; ni < 4; ++ni) bfr[ni] = frag_ld(Bs + ni * 16 * LDT_S, LDT_S, lane);
#pragma unroll
    for (int mi = 0; mi < 2; ++mi) {
      v16h a = frag_ld(As + mi * 16 * LDT_S, LDT_S, lane);
#pragma unroll
      for (int ni = 0; ni < 4; ++ni)
        acc[mi][ni] = wmma_f16(a, bfr[ni], acc[mi][ni]);
    }
    __syncthreads();
  }

  const int lr = (lane >> 4) * 8;
  const int lc = lane & 15;
#pragma unroll
  for (int mi = 0; mi < 2; ++mi)
#pragma unroll
    for (int ni = 0; ni < 4; ++ni) {
      const int cb = n0 + wc * 64 + ni * 16 + lc;
#pragma unroll
      for (int v = 0; v < 8; ++v) {
        const int r = m0 + wr * 32 + mi * 16 + lr + v;
        if (r < M && cb < N) {
          float x = alpha * acc[mi][ni][v];
          if (beta != 0.0f) x += beta * Cin[(size_t)r * ldc + cb];
          if (bias) x += bias[cb];
          if (act) x = gelu_f(x);
          if (outF) outF[(size_t)r * ldc + cb] = x;
          if (outH) outH[(size_t)r * ldc + cb] = (_Float16)x;
        }
      }
    }
}

// ------------------------- small elementwise kernels -----------------------

__global__ void k_twid(_Float16* Wr, _Float16* Wi, _Float16* Vr, _Float16* Vi) {
  long long idx = (long long)blockIdx.x * 256 + threadIdx.x;
  if (idx >= (long long)NS * NS) return;
  int n = (int)(idx / NS), k = (int)(idx % NS);
  int m = (int)(((long long)n * k) % NS);
  float th = 6.28318530718f * (float)m / (float)NS;
  float s, c;
  __sincosf(th, &s, &c);
  Wr[idx] = (_Float16)c;  Wi[idx] = (_Float16)(-s);
  Vr[idx] = (_Float16)(c * (1.0f / NS));  Vi[idx] = (_Float16)(s * (1.0f / NS));
}

__global__ void k_pack(const float* __restrict__ x, _Float16* Ar, _Float16* Ai) {
  long long idx = (long long)blockIdx.x * 256 + threadIdx.x;
  if (idx >= (long long)NB * NCO * NS) return;
  int n = (int)(idx % NS);
  int ro = (int)(idx / NS);
  int b = ro >> 3, o = ro & 7;
  size_t base = ((size_t)b * NS + n) * (2 * NCO) + 2 * o;
  Ar[idx] = (_Float16)x[base];
  Ai[idx] = (_Float16)x[base + 1];
}

__global__ void k_solve(const int* __restrict__ band, float* __restrict__ G) {
  int b = blockIdx.x;
  if (threadIdx.x != 0) return;
  float Ar[8][4], Ai[8][4];
  for (int o = 0; o < 8; ++o)
    for (int k = 0; k < 4; ++k) {
      int r = band[b * 4 + k];
      int m = (o * r) % RR;
      float th = 6.28318530718f * (float)m / (float)RR;
      Ar[o][k] = __cosf(th); Ai[o][k] = __sinf(th);
    }
  float Mr[4][4], Mi[4][4], Ir[4][4] = {}, Ii[4][4] = {};
  for (int k = 0; k < 4; ++k)
    for (int l = 0; l < 4; ++l) {
      float sr = 0, si = 0;
      for (int o = 0; o < 8; ++o) {
        sr += Ar[o][k] * Ar[o][l] + Ai[o][k] * Ai[o][l];
        si += Ar[o][k] * Ai[o][l] - Ai[o][k] * Ar[o][l];
      }
      Mr[k][l] = sr; Mi[k][l] = si;
    }
  for (int k = 0; k < 4; ++k) Ir[k][k] = 1.0f;
  for (int p = 0; p < 4; ++p) {
    float dr = Mr[p][p], di = Mi[p][p];
    float den = 1.0f / (dr * dr + di * di + 1e-20f);
    float pr = dr * den, pi = -di * den;
    for (int c = 0; c < 4; ++c) {
      float xr = Mr[p][c], xi = Mi[p][c];
      Mr[p][c] = xr * pr - xi * pi; Mi[p][c] = xr * pi + xi * pr;
      xr = Ir[p][c]; xi = Ii[p][c];
      Ir[p][c] = xr * pr - xi * pi; Ii[p][c] = xr * pi + xi * pr;
    }
    for (int rr2 = 0; rr2 < 4; ++rr2)
      if (rr2 != p) {
        float fr = Mr[rr2][p], fi = Mi[rr2][p];
        for (int c = 0; c < 4; ++c) {
          Mr[rr2][c] -= fr * Mr[p][c] - fi * Mi[p][c];
          Mi[rr2][c] -= fr * Mi[p][c] + fi * Mr[p][c];
          Ir[rr2][c] -= fr * Ir[p][c] - fi * Ii[p][c];
          Ii[rr2][c] -= fr * Ii[p][c] + fi * Ir[p][c];
        }
      }
  }
  for (int k = 0; k < 4; ++k)
    for (int o = 0; o < 8; ++o) {
      float sr = 0, si = 0;
      for (int l = 0; l < 4; ++l) {
        sr += Ir[k][l] * Ar[o][l] + Ii[k][l] * Ai[o][l];
        si += Ii[k][l] * Ar[o][l] - Ir[k][l] * Ai[o][l];
      }
      G[((b * 4 + k) * 8 + o) * 2 + 0] = sr;
      G[((b * 4 + k) * 8 + o) * 2 + 1] = si;
    }
}

__global__ void k_apply(const float* __restrict__ Yre, const float* __restrict__ Yim,
                        const float* __restrict__ G,
                        _Float16* __restrict__ Xr, _Float16* __restrict__ Xi) {
  long long idx = (long long)blockIdx.x * 256 + threadIdx.x;
  if (idx >= (long long)NB * NS) return;
  int b = (int)(idx / NS), t = (int)(idx % NS);
  float zr[8], zi[8];
  for (int o = 0; o < 8; ++o) {
    long long m = ((long long)o * t) % (RR * NS);
    float th = -6.28318530718f * (float)m / (float)(RR * NS);
    float c = __cosf(th), s = __sinf(th);
    size_t yb = ((size_t)(b * 8 + o)) * NS + t;
    float yr = Yre[yb], yi = Yim[yb];
    zr[o] = c * yr - s * yi;
    zi[o] = c * yi + s * yr;
  }
  for (int k = 0; k < 4; ++k) {
    float sr = 0, si = 0;
    for (int o = 0; o < 8; ++o) {
      float gr = G[((b * 4 + k) * 8 + o) * 2];
      float gi = G[((b * 4 + k) * 8 + o) * 2 + 1];
      sr += gr * zr[o] - gi * zi[o];
      si += gr * zi[o] + gi * zr[o];
    }
    Xr[((size_t)(b * 4 + k)) * NS + t] = (_Float16)sr;
    Xi[((size_t)(b * 4 + k)) * NS + t] = (_Float16)si;
  }
}

// unbiased std over (2400,2) row, then pack to (75,64) token features (f16)
__global__ void k_std(const float* __restrict__ Sr, const float* __restrict__ Si,
                      _Float16* __restrict__ zh) {
  __shared__ float red[256], red2[256];
  int r = blockIdx.x, tid = threadIdx.x;
  float s = 0, ss = 0;
  for (int t = tid; t < NS; t += 256) {
    float a = Sr[(size_t)r * NS + t], b2 = Si[(size_t)r * NS + t];
    s += a + b2; ss += a * a + b2 * b2;
  }
  red[tid] = s; red2[tid] = ss;
  __syncthreads();
  for (int o = 128; o > 0; o >>= 1) {
    if (tid < o) { red[tid] += red[tid + o]; red2[tid] += red2[tid + o]; }
    __syncthreads();
  }
  const float n = 2.0f * NS;
  float mean = red[0] / n;
  float var = (red2[0] - n * mean * mean) / (n - 1.0f);
  float inv = rsqrtf(var + 1e-20f);
  for (int t = tid; t < NS; t += 256) {
    int tok = t >> 5, j = t & 31;
    size_t base = (size_t)r * (TOK * 64) + (size_t)tok * 64 + j * 2;
    zh[base]     = (_Float16)(Sr[(size_t)r * NS + t] * inv);
    zh[base + 1] = (_Float16)(Si[(size_t)r * NS + t] * inv);
  }
}

// out = LN(a [+ res]) * g + b   (one block per row)
__global__ void k_ln(const _Float16* __restrict__ a, const _Float16* __restrict__ res,
                     const float* __restrict__ g, const float* __restrict__ bta,
                     _Float16* __restrict__ out, int n) {
  __shared__ float red[256];
  size_t row = blockIdx.x;
  int tid = threadIdx.x;
  float s = 0;
  for (int c = tid; c < n; c += 256) {
    float v = (float)a[row * n + c];
    if (res) v += (float)res[row * n + c];
    s += v;
  }
  red[tid] = s; __syncthreads();
  for (int o = 128; o > 0; o >>= 1) { if (tid < o) red[tid] += red[tid + o]; __syncthreads(); }
  float mean = red[0] / n;
  __syncthreads();
  float ss = 0;
  for (int c = tid; c < n; c += 256) {
    float v = (float)a[row * n + c];
    if (res) v += (float)res[row * n + c];
    v -= mean; ss += v * v;
  }
  red[tid] = ss; __syncthreads();
  for (int o = 128; o > 0; o >>= 1) { if (tid < o) red[tid] += red[tid + o]; __syncthreads(); }
  float inv = rsqrtf(red[0] / n + 1e-5f);
  for (int c = tid; c < n; c += 256) {
    float v = (float)a[row * n + c];
    if (res) v += (float)res[row * n + c];
    out[row * n + c] = (_Float16)((v - mean) * inv * g[c] + bta[c]);
  }
}

// f32 (KxN) -> f16 transposed (NxK): downstream B operands are row-contig in K
__global__ void k_cvtT(const float* __restrict__ s, _Float16* __restrict__ d,
                       int Kd, int N) {
  long long i = (long long)blockIdx.x * 256 + threadIdx.x;
  if (i >= (long long)Kd * N) return;
  int n = (int)(i / Kd), k = (int)(i % Kd);
  d[i] = (_Float16)s[(size_t)k * N + n];
}

__global__ void k_bcastq(const float* __restrict__ q, _Float16* __restrict__ out) {
  int i = blockIdx.x * 256 + threadIdx.x;
  if (i < 1024 * DD) out[i] = (_Float16)q[i % DD];
}

__global__ void k_addpos(const float* __restrict__ pos, _Float16* __restrict__ h) {
  long long i = (long long)blockIdx.x * 256 + threadIdx.x;
  if (i >= (long long)1024 * TOK * DD) return;
  int d = (int)(i % DD);
  int tok = (int)((i / DD) % TOK);
  h[i] = (_Float16)((float)h[i] + pos[tok * DD + d]);
}

// ----------------------- fused encoder attention ---------------------------
// One block per (sequence, head). WMMA for QK^T and PV; both B operands kept
// "transposed" in LDS so every fragment load is contiguous ds_load_b128.

__launch_bounds__(256)
__global__ void k_attn(const _Float16* __restrict__ qkv, _Float16* __restrict__ outb) {
  const int b = blockIdx.x, h = blockIdx.y;
  extern __shared__ __align__(16) char sm[];
  _Float16* Qs = (_Float16*)sm;              // 80 x 136  (rows = queries)
  _Float16* Ks = Qs + 80 * 136;              // 80 x 136  (rows = keys; == B^T)
  _Float16* Vt = Ks + 80 * 136;              // 128 x 104 (rows = dims; == V^T)
  float*    Ps = (float*)(Vt + 128 * 104);   // 80 x 88  scores f32
  _Float16* Pf = (_Float16*)(Ps + 80 * 88);  // 80 x 104 probs f16
  const int tid = threadIdx.x, lane = tid & 31, wid = tid >> 5;

  for (int i = tid; i < 80 * 128; i += 256) {
    int r = i >> 7, d = i & 127;
    float q = 0, k = 0, v = 0;
    if (r < TOK) {
      size_t base = ((size_t)(b * TOK + r)) * (3 * DD) + (size_t)h * DH + d;
      q = (float)qkv[base];
      k = (float)qkv[base + DD];
      v = (float)qkv[base + 2 * DD];
    }
    Qs[r * 136 + d] = (_Float16)q;
    Ks[r * 136 + d] = (_Float16)k;
    Vt[d * 104 + r] = (_Float16)v;
  }
  for (int i = tid; i < 128 * 16; i += 256) {        // Vt cols 80..95 = 0
    int d = i >> 4, j = 80 + (i & 15);
    Vt[d * 104 + j] = (_Float16)0.0f;
  }
  __syncthreads();

  // scores = Q K^T / sqrt(dh)   (5x5 tiles of 16x16, K=128)
  for (int t = wid; t < 25; t += 8) {
    int ti = t / 5, tj = t % 5;
    v8f acc = {};
#pragma unroll
    for (int kb = 0; kb < 128; kb += 32) {
      v16h a = frag_ld(Qs + ti * 16 * 136 + kb, 136, lane);
      v16h bf = frag_ld(Ks + tj * 16 * 136 + kb, 136, lane);
      acc = wmma_f16(a, bf, acc);
    }
    int lr = (lane >> 4) * 8, lc = lane & 15;
#pragma unroll
    for (int v = 0; v < 8; ++v)
      Ps[(ti * 16 + lr + v) * 88 + tj * 16 + lc] = acc[v] * 0.0883883476f;
  }
  __syncthreads();

  if (tid < 80) {
    int r = tid;
    if (r < TOK) {
      float mx = -1e30f;
      for (int c = 0; c < TOK; ++c) mx = fmaxf(mx, Ps[r * 88 + c]);
      float s = 0;
      for (int c = 0; c < TOK; ++c) { float e = __expf(Ps[r * 88 + c] - mx); Ps[r * 88 + c] = e; s += e; }
      float inv = 1.0f / s;
      for (int c = 0; c < 96; ++c)
        Pf[r * 104 + c] = (_Float16)(c < TOK ? Ps[r * 88 + c] * inv : 0.0f);
    } else {
      for (int c = 0; c < 96; ++c) Pf[r * 104 + c] = (_Float16)0.0f;
    }
  }
  __syncthreads();

  // out = P V   (5x8 tiles, contraction padded to 96; Vt rows are output dims)
  for (int t = wid; t < 40; t += 8) {
    int ti = t >> 3, dj = t & 7;
    v8f acc = {};
#pragma unroll
    for (int kb = 0; kb < 96; kb += 32) {
      v16h a = frag_ld(Pf + ti * 16 * 104 + kb, 104, lane);
      v16h bf = frag_ld(Vt + dj * 16 * 104 + kb, 104, lane);
      acc = wmma_f16(a, bf, acc);
    }
    int lr = (lane >> 4) * 8, lc = lane & 15;
#pragma unroll
    for (int v = 0; v < 8; ++v) {
      int r = ti * 16 + lr + v;
      if (r < TOK)
        outb[((size_t)(b * TOK + r)) * DD + (size_t)h * DH + dj * 16 + lc] = (_Float16)acc[v];
    }
  }
}

// Decoder cross-attention: single query vs 75 memory tokens (VALU; tiny).
__global__ void k_xattn(const _Float16* __restrict__ qd, const _Float16* __restrict__ kv,
                        _Float16* __restrict__ outb) {
  int b = blockIdx.x, tid = threadIdx.x;
  __shared__ float sc[NHD][80], pr[NHD][80], qsh[DD];
  for (int i = tid; i < DD; i += 256) qsh[i] = (float)qd[(size_t)b * DD + i];
  __syncthreads();
  for (int i = tid; i < NHD * TOK; i += 256) {
    int hh = i / TOK, j = i % TOK;
    const _Float16* kp = kv + ((size_t)(b * TOK + j)) * (2 * DD) + (size_t)hh * DH;
    float s = 0;
    for (int d = 0; d < DH; ++d) s += qsh[hh * DH + d] * (float)kp[d];
    sc[hh][j] = s * 0.0883883476f;
  }
  __syncthreads();
  if (tid < NHD) {
    float mx = -1e30f;
    for (int j = 0; j < TOK; ++j) mx = fmaxf(mx, sc[tid][j]);
    float s = 0;
    for (int j = 0; j < TOK; ++j) { float e = __expf(sc[tid][j] - mx); pr[tid][j] = e; s += e; }
    float inv = 1.0f / s;
    for (int j = 0; j < TOK; ++j) pr[tid][j] *= inv;
  }
  __syncthreads();
  for (int i = tid; i < DD; i += 256) {
    int hh = i >> 7, d = i & 127;
    float o = 0;
    for (int j = 0; j < TOK; ++j)
      o += pr[hh][j] * (float)kv[((size_t)(b * TOK + j)) * (2 * DD) + DD + (size_t)hh * DH + d];
    outb[(size_t)b * DD + i] = (_Float16)o;
  }
}

// ------------------------------ host launcher ------------------------------

static void gemm(hipStream_t st, const _Float16* A, int lda, const _Float16* Bt, int ldbt,
                 const float* bias, const float* Cin, float* outF, _Float16* outH,
                 int ldc, int M, int N, int K, float alpha, float beta, int act) {
  dim3 g((N + GBN - 1) / GBN, (M + GBM - 1) / GBM);
  k_gemm<<<g, 256, 0, st>>>(A, lda, Bt, ldbt, bias, Cin, outF, outH, ldc, M, N, K, alpha, beta, act);
}

extern "C" void kernel_launch(void* const* d_in, const int* in_sizes, int n_in,
                              void* d_out, int out_size, void* d_ws, size_t ws_size,
                              hipStream_t stream) {
  (void)in_sizes; (void)n_in; (void)out_size; (void)ws_size;
  const float* x    = (const float*)d_in[0];
  const int*   band = (const int*)d_in[1];
  auto F = [&](int i) { return (const float*)d_in[i]; };

  // workspace bump allocator
  char* W = (char*)d_ws;
  size_t off = 0;
  auto alloc = [&](size_t bytes) -> void* {
    void* r = W + off;
    off = (off + bytes + 255) & ~(size_t)255;
    return r;
  };
  const size_t H = sizeof(_Float16);

  // DFT twiddle matrices are symmetric: they equal their own transpose, so
  // they serve directly as Bt (NxK) operands.
  _Float16* Wr = (_Float16*)alloc((size_t)NS * NS * H);
  _Float16* Wi = (_Float16*)alloc((size_t)NS * NS * H);
  _Float16* Vr = (_Float16*)alloc((size_t)NS * NS * H);
  _Float16* Vi = (_Float16*)alloc((size_t)NS * NS * H);
  _Float16* Arh = (_Float16*)alloc((size_t)NB * NCO * NS * H);
  _Float16* Aih = (_Float16*)alloc((size_t)NB * NCO * NS * H);
  float* Yre = (float*)alloc((size_t)NB * NCO * NS * 4);
  float* Yim = (float*)alloc((size_t)NB * NCO * NS * 4);
  float* Gm  = (float*)alloc((size_t)NB * NK * NCO * 2 * 4);
  _Float16* Xrh = (_Float16*)alloc((size_t)NB * NK * NS * H);
  _Float16* Xih = (_Float16*)alloc((size_t)NB * NK * NS * H);
  float* Srf = (float*)alloc((size_t)NB * NK * NS * 4);
  float* Sif = (float*)alloc((size_t)NB * NK * NS * 4);
  _Float16* zh = (_Float16*)alloc((size_t)1024 * TOK * 64 * H);

  // f16 transposed weights (NxK)
  _Float16* sw1t = (_Float16*)alloc((size_t)64 * 1024 * H);
  _Float16* sw2t = (_Float16*)alloc((size_t)1024 * 1024 * H);
  _Float16* sw3t = (_Float16*)alloc((size_t)1024 * 512 * H);
  _Float16 *eint[NLAY], *eoutt[NLAY], *ef1t[NLAY], *ef2t[NLAY];
  for (int e = 0; e < NLAY; ++e) {
    eint[e]  = (_Float16*)alloc((size_t)512 * 1536 * H);
    eoutt[e] = (_Float16*)alloc((size_t)512 * 512 * H);
    ef1t[e]  = (_Float16*)alloc((size_t)512 * 1536 * H);
    ef2t[e]  = (_Float16*)alloc((size_t)1536 * 512 * H);
  }
  _Float16* saint  = (_Float16*)alloc((size_t)512 * 1536 * H);
  _Float16* saoutt = (_Float16*)alloc((size_t)512 * 512 * H);
  _Float16* caint  = (_Float16*)alloc((size_t)512 * 1536 * H);
  _Float16* caoutt = (_Float16*)alloc((size_t)512 * 512 * H);
  _Float16* df1t   = (_Float16*)alloc((size_t)512 * 1536 * H);
  _Float16* df2t   = (_Float16*)alloc((size_t)1536 * 512 * H);
  _Float16* fw1t   = (_Float16*)alloc((size_t)512 * 512 * H);
  _Float16* fw2t   = (_Float16*)alloc((size_t)512 * NCLS * H);

  // big activation arena (reused across stages)
  const size_t MTOK = (size_t)1024 * TOK;  // 76800
  char* arena = (char*)alloc(MTOK * 1024 * H * 2);  // 314.6 MB
  _Float16* g1buf = (_Float16*)arena;                    // 76800x1024
  _Float16* l1buf = (_Float16*)(arena + MTOK * 1024 * H);
  _Float16* qkvp  = (_Float16*)arena;                    // 76800x1536
  _Float16* aop   = (_Float16*)(arena + MTOK * 1536 * H);
  _Float16* ff1p  = (_Float16*)arena;                    // 76800x1536
  _Float16* kvp   = (_Float16*)arena;                    // 76800x1024
  _Float16* hA    = (_Float16*)alloc(MTOK * DD * H);
  _Float16* hB    = (_Float16*)alloc(MTOK * DD * H);
  _Float16* attnb = (_Float16*)alloc(MTOK * DD * H);
  _Float16* sqkv  = (_Float16*)alloc((size_t)1024 * 1536 * H);
  _Float16* q0h   = (_Float16*)alloc((size_t)1024 * DD * H);
  _Float16* q1h   = (_Float16*)alloc((size_t)1024 * DD * H);
  _Float16* q2h   = (_Float16*)alloc((size_t)1024 * DD * H);
  _Float16* q3h   = (_Float16*)alloc((size_t)1024 * DD * H);
  _Float16* qdh   = (_Float16*)alloc((size_t)1024 * DD * H);
  _Float16* caoh  = (_Float16*)alloc((size_t)1024 * DD * H);
  _Float16* tmp5  = (_Float16*)alloc((size_t)1024 * DD * H);
  _Float16* dffb  = (_Float16*)alloc((size_t)1024 * 1536 * H);
  _Float16* fct   = (_Float16*)alloc((size_t)1024 * DD * H);

  auto cvtT = [&](const float* s, _Float16* d, int K, int N) {
    long long n = (long long)K * N;
    k_cvtT<<<(unsigned)((n + 255) / 256), 256, 0, stream>>>(s, d, K, N);
  };

  // ---- weight conversion (transpose to NxK) ----
  cvtT(F(2), sw1t, 64, 1024);
  cvtT(F(6), sw2t, 1024, 1024);
  cvtT(F(10), sw3t, 1024, 512);
  for (int e = 0; e < NLAY; ++e) {
    int p = 16 + 12 * e;
    cvtT(F(p + 0), eint[e], 512, 1536);
    cvtT(F(p + 2), eoutt[e], 512, 512);
    cvtT(F(p + 6), ef1t[e], 512, 1536);
    cvtT(F(p + 8), ef2t[e], 1536, 512);
  }
  cvtT(F(64), saint, 512, 1536);
  cvtT(F(66), saoutt, 512, 512);
  cvtT(F(68), caint, 512, 1536);
  cvtT(F(70), caoutt, 512, 512);
  cvtT(F(72), df1t, 512, 1536);
  cvtT(F(74), df2t, 1536, 512);
  cvtT(F(82), fw1t, 512, 512);
  cvtT(F(86), fw2t, 512, NCLS);

  // ---- signal recovery ----
  k_twid<<<((NS * NS) + 255) / 256, 256, 0, stream>>>(Wr, Wi, Vr, Vi);
  k_pack<<<((NB * NCO * NS) + 255) / 256, 256, 0, stream>>>(x, Arh, Aih);
  const int MR = NB * NCO;  // 2048
  gemm(stream, Arh, NS, Wr, NS, nullptr, nullptr, Yre, nullptr, NS, MR, NS, NS, 1.f, 0.f, 0);
  gemm(stream, Aih, NS, Wi, NS, nullptr, Yre, Yre, nullptr, NS, MR, NS, NS, -1.f, 1.f, 0);
  gemm(stream, Arh, NS, Wi, NS, nullptr, nullptr, Yim, nullptr, NS, MR, NS, NS, 1.f, 0.f, 0);
  gemm(stream, Aih, NS, Wr, NS, nullptr, Yim, Yim, nullptr, NS, MR, NS, NS, 1.f, 1.f, 0);
  k_solve<<<NB, 32, 0, stream>>>(band, Gm);
  k_apply<<<((NB * NS) + 255) / 256, 256, 0, stream>>>(Yre, Yim, Gm, Xrh, Xih);
  const int MI = NB * NK;  // 1024
  gemm(stream, Xrh, NS, Vr, NS, nullptr, nullptr, Srf, nullptr, NS, MI, NS, NS, 1.f, 0.f, 0);
  gemm(stream, Xih, NS, Vi, NS, nullptr, Srf, Srf, nullptr, NS, MI, NS, NS, -1.f, 1.f, 0);
  gemm(stream, Xrh, NS, Vi, NS, nullptr, nullptr, Sif, nullptr, NS, MI, NS, NS, 1.f, 0.f, 0);
  gemm(stream, Xih, NS, Vr, NS, nullptr, Sif, Sif, nullptr, NS, MI, NS, NS, 1.f, 1.f, 0);
  k_std<<<MI, 256, 0, stream>>>(Srf, Sif, zh);

  // ---- slot MLP ----
  const int MT = (int)MTOK;  // 76800
  gemm(stream, zh, 64, sw1t, 64, F(3), nullptr, nullptr, g1buf, 1024, MT, 1024, 64, 1.f, 0.f, 1);
  k_ln<<<MT, 256, 0, stream>>>(g1buf, nullptr, F(4), F(5), l1buf, 1024);
  gemm(stream, l1buf, 1024, sw2t, 1024, F(7), nullptr, nullptr, g1buf, 1024, MT, 1024, 1024, 1.f, 0.f, 1);
  k_ln<<<MT, 256, 0, stream>>>(g1buf, nullptr, F(8), F(9), l1buf, 1024);
  gemm(stream, l1buf, 1024, sw3t, 1024, F(11), nullptr, nullptr, hA, 512, MT, 512, 1024, 1.f, 0.f, 0);
  k_ln<<<MT, 256, 0, stream>>>(hA, nullptr, F(12), F(13), hA, 512);
  k_addpos<<<(unsigned)(((long long)MT * 512 + 255) / 256), 256, 0, stream>>>(F(14), hA);

  // ---- encoder layers ----
  const size_t attn_lds = 80 * 136 * H * 2 + 128 * 104 * H + 80 * 88 * 4 + 80 * 104 * H;
  for (int e = 0; e < NLAY; ++e) {
    int p = 16 + 12 * e;
    gemm(stream, hA, 512, eint[e], 512, F(p + 1), nullptr, nullptr, qkvp, 1536, MT, 1536, 512, 1.f, 0.f, 0);
    k_attn<<<dim3(1024, NHD), 256, attn_lds, stream>>>(qkvp, attnb);
    gemm(stream, attnb, 512, eoutt[e], 512, F(p + 3), nullptr, nullptr, aop, 512, MT, 512, 512, 1.f, 0.f, 0);
    k_ln<<<MT, 256, 0, stream>>>(hA, aop, F(p + 4), F(p + 5), hB, 512);
    gemm(stream, hB, 512, ef1t[e], 512, F(p + 7), nullptr, nullptr, ff1p, 1536, MT, 1536, 512, 1.f, 0.f, 1);
    gemm(stream, ff1p, 1536, ef2t[e], 1536, F(p + 9), nullptr, nullptr, attnb, 512, MT, 512, 1536, 1.f, 0.f, 0);
    k_ln<<<MT, 256, 0, stream>>>(hB, attnb, F(p + 10), F(p + 11), hA, 512);
  }

  // ---- decoder (single query) ----
  k_bcastq<<<(1024 * DD + 255) / 256, 256, 0, stream>>>(F(15), q0h);
  // self-attn over seq-len 1: softmax==1 -> out = V = q0@Wv+bv, then out proj
  gemm(stream, q0h, 512, saint, 512, F(65), nullptr, nullptr, sqkv, 1536, 1024, 1536, 512, 1.f, 0.f, 0);
  gemm(stream, sqkv + 1024, 1536, saoutt, 512, F(67), nullptr, nullptr, tmp5, 512, 1024, 512, 512, 1.f, 0.f, 0);
  k_ln<<<1024, 256, 0, stream>>>(q0h, tmp5, F(76), F(77), q1h, 512);
  // cross-attn: K,V from encoder memory (rows 512..1535 of transposed in_w)
  gemm(stream, hA, 512, caint + (size_t)512 * 512, 512, F(69) + 512, nullptr, nullptr, kvp, 1024,
       MT, 1024, 512, 1.f, 0.f, 0);
  gemm(stream, q1h, 512, caint, 512, F(69), nullptr, nullptr, qdh, 512, 1024, 512, 512, 1.f, 0.f, 0);
  k_xattn<<<1024, 256, 0, stream>>>(qdh, kvp, caoh);
  gemm(stream, caoh, 512, caoutt, 512, F(71), nullptr, nullptr, tmp5, 512, 1024, 512, 512, 1.f, 0.f, 0);
  k_ln<<<1024, 256, 0, stream>>>(q1h, tmp5, F(78), F(79), q2h, 512);
  gemm(stream, q2h, 512, df1t, 512, F(73), nullptr, nullptr, dffb, 1536, 1024, 1536, 512, 1.f, 0.f, 1);
  gemm(stream, dffb, 1536, df2t, 1536, F(75), nullptr, nullptr, tmp5, 512, 1024, 512, 1536, 1.f, 0.f, 0);
  k_ln<<<1024, 256, 0, stream>>>(q2h, tmp5, F(80), F(81), q3h, 512);

  // ---- fc head ----
  gemm(stream, q3h, 512, fw1t, 512, F(83), nullptr, nullptr, fct, 512, 1024, 512, 512, 1.f, 0.f, 1);
  k_ln<<<1024, 256, 0, stream>>>(fct, nullptr, F(84), F(85), q0h, 512);
  gemm(stream, q0h, 512, fw2t, 512, F(87), nullptr, (float*)d_out, nullptr, NCLS,
       1024, NCLS, 512, 1.f, 0.f, 0);
}
